// GraphTransformerLayer_17901423689811
// MI455X (gfx1250) — matrix operand order repaired and checked by reading, not compiled, not run on hardware
//
#include <hip/hip_runtime.h>
#include <hip/hip_bf16.h>

#define NNODES 1024
#define DDIM   128
#define NHEAD  8
#define HDIM   16
#define TOPK   16

typedef __attribute__((ext_vector_type(16))) __bf16 v16bf;
typedef __attribute__((ext_vector_type(8)))  float  v8f;

union Frag {
  v16bf v;
  unsigned short s[16];
  uint4 q[2];
};

// Native convert: clang lowers fptrunc f32->bf16 to the gfx1250 cvt ops.
__device__ __forceinline__ unsigned short bfbits(float f) {
  union { __bf16 b; unsigned short u; } x;
  x.b = (__bf16)f;
  return x.u;
}

// Build a 16-element bf16 A/B fragment from 4 float4 loads.
__device__ __forceinline__ v16bf cvt16(float4 f0, float4 f1, float4 f2, float4 f3) {
  v16bf r;
  r[0]  = (__bf16)f0.x; r[1]  = (__bf16)f0.y; r[2]  = (__bf16)f0.z; r[3]  = (__bf16)f0.w;
  r[4]  = (__bf16)f1.x; r[5]  = (__bf16)f1.y; r[6]  = (__bf16)f1.z; r[7]  = (__bf16)f1.w;
  r[8]  = (__bf16)f2.x; r[9]  = (__bf16)f2.y; r[10] = (__bf16)f2.z; r[11] = (__bf16)f2.w;
  r[12] = (__bf16)f3.x; r[13] = (__bf16)f3.y; r[14] = (__bf16)f3.z; r[15] = (__bf16)f3.w;
  return r;
}

// Per-lane A fragment from a row-major fp32 matrix: lane row = lane&15,
// K runs [kb*32+16*hi, +16) (contiguous order, matches packed B).
__device__ __forceinline__ v16bf load_frag_a(const float* __restrict__ arow, int kb, int hi) {
  const float4* p1 = (const float4*)(arow + kb * 32 + 16 * hi);
  return cvt16(p1[0], p1[1], p1[2], p1[3]);
}

// ---------------------------------------------------------------------------
// Pack an fp32 weight matrix (B operand, [Kd x Nn] row-major, or transposed
// source W[Nn x Kd] when trans=1) into per-lane WMMA bf16 B-fragment order:
// layout index = ((kb*NT + nt)*32 + lane)*16 + e, with K = kb*32 + e + 16*hi.
// ---------------------------------------------------------------------------
__global__ void pack_b_kernel(const float* __restrict__ W, unsigned short* __restrict__ out,
                              int Kd, int Nn, int NT, int trans) {
  int t = blockIdx.x * blockDim.x + threadIdx.x;
  int KB = Kd >> 5;
  if (t >= KB * NT * 32) return;
  int lane = t & 31;
  int tile = t >> 5;
  int nt = tile % NT;
  int kb = tile / NT;
  int hi = lane >> 4;
  int n  = nt * 16 + (lane & 15);
  unsigned short* o = out + (size_t)t * 16;
#pragma unroll
  for (int e = 0; e < 16; ++e) {
    int kk = kb * 32 + e + 16 * hi;
    float val = 0.0f;
    if (kk < Kd && n < Nn)
      val = trans ? W[(size_t)n * Kd + kk] : W[(size_t)kk * Nn + n];
    o[e] = bfbits(val);
  }
}

// ---------------------------------------------------------------------------
// Generic GEMM: C[M x Nn] = A[M x KD](fp32 -> bf16 on the fly) * Bpacked
//               + bias (+bias2), optional ReLU. One wave per 16x16 C tile.
// KD templated so the K loop fully unrolls into straight-line WMMAs.
// ---------------------------------------------------------------------------
template <int KD>
__global__ void gemm_bf16_wmma(const float* __restrict__ A,
                               const unsigned short* __restrict__ Bp,
                               const float* __restrict__ bias,
                               const float* __restrict__ bias2,
                               float* __restrict__ C,
                               int M, int Nn, int NT, int act) {
  const int lane = threadIdx.x & 31;
  const int wave = threadIdx.x >> 5;
  const int tile = blockIdx.x * (blockDim.x >> 5) + wave;
  const int tilesM = M >> 4;
  if (tile >= tilesM * NT) return;
  const int mt = tile / NT;
  const int nt = tile - mt * NT;
  const int m  = lane & 15;
  const int hi = lane >> 4;

  const float* arow = A + (size_t)(mt * 16 + m) * KD;
  __builtin_prefetch(arow + (size_t)16 * KD, 0, 1);   // next M-tile, same lane row

  v8f acc = {};
#pragma unroll
  for (int kb = 0; kb < (KD >> 5); ++kb) {
    Frag b;
    const uint4* bp = (const uint4*)(Bp + (((size_t)(kb * NT + nt) * 32 + lane) << 4));
    b.q[0] = bp[0];
    b.q[1] = bp[1];
    v16bf a = load_frag_a(arow, kb, hi);
    acc = __builtin_amdgcn_wmma_f32_16x16x32_bf16(false, a, false, b.v,
                                                  (short)0, acc, false, false);
  }

  const int n = nt * 16 + (lane & 15);
  if (n < Nn) {
    float bsum = (bias ? bias[n] : 0.0f) + (bias2 ? bias2[n] : 0.0f);
#pragma unroll
    for (int r = 0; r < 8; ++r) {
      float vo = acc[r] + bsum;
      if (act) vo = fmaxf(vo, 0.0f);
      C[(size_t)(mt * 16 + 8 * hi + r) * Nn + n] = vo;
    }
  }
}

// ---------------------------------------------------------------------------
// Fused score kernel: for each 16-row tile of flattened (i,j) pairs
// (all 16 rows share one i), compute
//   scores[j, h] = e[i,j,:]@We[:,h] + be[h] + (q[i,h,:].k[j,h,:]) / 4
// The qk term is a second WMMA per K-block with B_q[kk, h] = 0.25*q[i,kk]
// masked to kk in head h: for K-block kb, the nonzero B columns are heads
// 2kb (lanes hi=0) and 2kb+1 (lanes hi=1) -> one 64B q load per half-wave.
// Streams the 512 MB e tensor exactly once; 8 WMMAs per tile.
// ---------------------------------------------------------------------------
__global__ void escore_kernel(const float* __restrict__ eg,
                              const float* __restrict__ qg,
                              const float* __restrict__ kg,
                              const unsigned short* __restrict__ pWe,
                              const float* __restrict__ be,
                              float* __restrict__ sc) {
  const int lane = threadIdx.x & 31;
  const int wave = threadIdx.x >> 5;
  const int tile = blockIdx.x * 8 + wave;           // (N*N)/16 tiles
  if (tile >= (NNODES * NNODES) / 16) return;
  const int m  = lane & 15;
  const int hi = lane >> 4;
  const int n  = lane & 15;                         // output column = head
  const int i  = (tile * 16) >> 10;                 // node i (constant in tile)

  const float* erow = eg + ((size_t)tile * 16 + m) * DDIM;
  const float* krow = kg + (size_t)((tile * 16) & (NNODES - 1)) * DDIM + (size_t)m * DDIM;
  const float* qrow = qg + (size_t)i * DDIM;
  __builtin_prefetch(erow + (size_t)16 * DDIM, 0, 1);

  v8f acc = {};
#pragma unroll
  for (int kb = 0; kb < 4; ++kb) {
    // --- e @ We contribution ---
    Frag bw;
    const uint4* bp = (const uint4*)(pWe + (((size_t)kb * 32 + lane) << 4));
    bw.q[0] = bp[0];
    bw.q[1] = bp[1];
    v16bf ae = load_frag_a(erow, kb, hi);
    acc = __builtin_amdgcn_wmma_f32_16x16x32_bf16(false, ae, false, bw.v,
                                                  (short)0, acc, false, false);
    // --- q.k^T contribution (scaled by 1/sqrt(16)) ---
    v16bf ak = load_frag_a(krow, kb, hi);
    v16bf bq = {};
    if (n == 2 * kb + hi) {                         // this column owns head n
      const float4* qp = (const float4*)(qrow + n * HDIM);
      float4 q0 = qp[0], q1 = qp[1], q2 = qp[2], q3 = qp[3];
      bq = cvt16(make_float4(0.25f * q0.x, 0.25f * q0.y, 0.25f * q0.z, 0.25f * q0.w),
                 make_float4(0.25f * q1.x, 0.25f * q1.y, 0.25f * q1.z, 0.25f * q1.w),
                 make_float4(0.25f * q2.x, 0.25f * q2.y, 0.25f * q2.z, 0.25f * q2.w),
                 make_float4(0.25f * q3.x, 0.25f * q3.y, 0.25f * q3.z, 0.25f * q3.w));
    }
    acc = __builtin_amdgcn_wmma_f32_16x16x32_bf16(false, ak, false, bq,
                                                  (short)0, acc, false, false);
  }

  if (n < NHEAD) {
    float bsum = be[n];
#pragma unroll
    for (int r = 0; r < 8; ++r)
      sc[(size_t)(tile * 16 + 8 * hi + r) * NHEAD + n] = acc[r] + bsum;
  }
}

// ---------------------------------------------------------------------------
// Per (i, head): top-16 over 1024 neighbors, softmax on selected, attn . v.
// One wave per (i, h). Scores row cached in 32 regs/lane; 16 argmax rounds
// via shfl_xor butterflies.
// ---------------------------------------------------------------------------
__global__ void topk_softmax_av_kernel(const float* __restrict__ sc,
                                       const float* __restrict__ vg,
                                       float* __restrict__ hattn) {
  const int lane = threadIdx.x & 31;
  const int wave = threadIdx.x >> 5;
  const int flat = blockIdx.x * (blockDim.x >> 5) + wave;
  if (flat >= NNODES * NHEAD) return;
  const int h = flat & 7;
  const int i = flat >> 3;

  float sv[32];
  const float* srow = sc + (size_t)i * NNODES * NHEAD + h;
#pragma unroll
  for (int t = 0; t < 32; ++t)
    sv[t] = srow[(size_t)(t * 32 + lane) * NHEAD];

  unsigned used = 0;
  int   selj[TOPK];
  float sele[TOPK];
  float vmax = 0.0f, ssum = 0.0f;

#pragma unroll
  for (int s = 0; s < TOPK; ++s) {
    float best = -3.0e38f;
    int bt = 0;
#pragma unroll
    for (int t = 0; t < 32; ++t)
      if (!((used >> t) & 1u) && sv[t] > best) { best = sv[t]; bt = t; }
    int bj = bt * 32 + lane;
#pragma unroll
    for (int off = 16; off; off >>= 1) {
      float ov = __shfl_xor(best, off, 32);
      int   oj = __shfl_xor(bj,   off, 32);
      if (ov > best || (ov == best && oj < bj)) { best = ov; bj = oj; }
    }
    if ((bj & 31) == lane) used |= 1u << (bj >> 5);
    if (s == 0) vmax = best;
    float ev = __expf(best - vmax);
    ssum += ev;
    selj[s] = bj;
    sele[s] = ev;
  }

  const float inv = 1.0f / ssum;
  if (lane < HDIM) {
    float acc = 0.0f;
#pragma unroll
    for (int s = 0; s < TOPK; ++s)
      acc += sele[s] * inv * vg[(size_t)selj[s] * DDIM + h * HDIM + lane];
    hattn[(size_t)i * DDIM + h * HDIM + lane] = acc;
  }
}

// ---------------------------------------------------------------------------
// out[row,:] = LN(a[row,:] + b[row,:]) with gamma/beta.  One block (128 thr)
// per row, population variance.
// ---------------------------------------------------------------------------
__global__ void add_layernorm_kernel(const float* __restrict__ a,
                                     const float* __restrict__ badd,
                                     const float* __restrict__ g,
                                     const float* __restrict__ bb,
                                     float* __restrict__ out) {
  const int row = blockIdx.x;
  const int tid = threadIdx.x;
  float x = a[(size_t)row * DDIM + tid] + badd[(size_t)row * DDIM + tid];
  __shared__ float s1[DDIM];
  __shared__ float s2[DDIM];
  s1[tid] = x; s2[tid] = x * x;
  __syncthreads();
  for (int off = 64; off; off >>= 1) {
    if (tid < off) { s1[tid] += s1[tid + off]; s2[tid] += s2[tid + off]; }
    __syncthreads();
  }
  float mu  = s1[0] * (1.0f / DDIM);
  float var = s2[0] * (1.0f / DDIM) - mu * mu;
  float r   = rsqrtf(var + 1e-5f);
  out[(size_t)row * DDIM + tid] = g[tid] * (x - mu) * r + bb[tid];
}

// ---------------------------------------------------------------------------
// LSTMCell (zero state): c = sig(gi)*tanh(gg); hn = sig(go)*tanh(c);
// x2 = x + hn; also emits h_next / c_next outputs.
// ---------------------------------------------------------------------------
__global__ void lstm_elem_kernel(const float* __restrict__ x,
                                 const float* __restrict__ gates,
                                 float* __restrict__ x2,
                                 float* __restrict__ hout,
                                 float* __restrict__ cout) {
  int idx = blockIdx.x * blockDim.x + threadIdx.x;
  if (idx >= NNODES * DDIM) return;
  int i = idx >> 7, d = idx & 127;
  const float* g = gates + (size_t)i * (4 * DDIM);
  float gi = g[d], gg = g[2 * DDIM + d], go = g[3 * DDIM + d];
  float ig = 1.0f / (1.0f + __expf(-gi));
  float gt = tanhf(gg);
  float og = 1.0f / (1.0f + __expf(-go));
  float c  = ig * gt;
  float hn = og * tanhf(c);
  x2[idx]   = x[idx] + hn;
  hout[idx] = hn;
  cout[idx] = c;
}

// ---------------------------------------------------------------------------
// GraphNorm: normalize each feature column over the 1024 nodes.
// One block (256 thr) per column.
// ---------------------------------------------------------------------------
__global__ void graphnorm_kernel(const float* __restrict__ x,
                                 const float* __restrict__ g,
                                 const float* __restrict__ b,
                                 float* __restrict__ out) {
  const int d = blockIdx.x;
  const int tid = threadIdx.x;
  float s = 0.0f, s2 = 0.0f;
  for (int i = tid; i < NNODES; i += 256) {
    float v = x[(size_t)i * DDIM + d];
    s += v; s2 += v * v;
  }
  __shared__ float r1[256];
  __shared__ float r2[256];
  r1[tid] = s; r2[tid] = s2;
  __syncthreads();
  for (int off = 128; off; off >>= 1) {
    if (tid < off) { r1[tid] += r1[tid + off]; r2[tid] += r2[tid + off]; }
    __syncthreads();
  }
  float mu  = r1[0] * (1.0f / NNODES);
  float var = r2[0] * (1.0f / NNODES) - mu * mu;
  float rr  = rsqrtf(var + 1e-5f);
  float gg = g[d], bb = b[d];
  for (int i = tid; i < NNODES; i += 256)
    out[(size_t)i * DDIM + d] = gg * (x[(size_t)i * DDIM + d] - mu) * rr + bb;
}

// ---------------------------------------------------------------------------
extern "C" void kernel_launch(void* const* d_in, const int* in_sizes, int n_in,
                              void* d_out, int out_size, void* d_ws, size_t ws_size,
                              hipStream_t stream) {
  (void)in_sizes; (void)n_in; (void)out_size; (void)ws_size;
  const float* h   = (const float*)d_in[0];
  const float* e   = (const float*)d_in[1];
  // d_in[2] = adjm_bin (unused, matching reference)
  const float* Wq  = (const float*)d_in[3];
  const float* bq  = (const float*)d_in[4];
  const float* Wk  = (const float*)d_in[5];
  const float* bk  = (const float*)d_in[6];
  const float* Wv  = (const float*)d_in[7];
  const float* bv  = (const float*)d_in[8];
  const float* We  = (const float*)d_in[9];
  const float* be  = (const float*)d_in[10];
  const float* Wo  = (const float*)d_in[11];
  const float* bo  = (const float*)d_in[12];
  const float* Wf1 = (const float*)d_in[13];
  const float* bf1 = (const float*)d_in[14];
  const float* Wf2 = (const float*)d_in[15];
  const float* bf2 = (const float*)d_in[16];
  const float* Wih = (const float*)d_in[17];
  // d_in[18] = Whh (multiplied by zero state; only bhh survives)
  const float* bih = (const float*)d_in[19];
  const float* bhh = (const float*)d_in[20];
  const float* gng = (const float*)d_in[21];
  const float* gnb = (const float*)d_in[22];
  const float* l1g = (const float*)d_in[23];
  const float* l1b = (const float*)d_in[24];
  const float* l2g = (const float*)d_in[25];
  const float* l2b = (const float*)d_in[26];

  char* ws = (char*)d_ws;
  size_t off = 0;
  auto alloc = [&](size_t bytes) -> char* {
    off = (off + 255) & ~(size_t)255;
    char* p = ws + off;
    off += bytes;
    return p;
  };

  const size_t ND = (size_t)NNODES * DDIM;
  float* qb     = (float*)alloc(ND * 4);
  float* kbuf   = (float*)alloc(ND * 4);
  float* vb     = (float*)alloc(ND * 4);
  float* scores = (float*)alloc((size_t)NNODES * NNODES * NHEAD * 4);
  float* hattn  = (float*)alloc(ND * 4);
  float* tmp    = (float*)alloc(ND * 4);
  float* xb     = (float*)alloc(ND * 4);
  float* gates  = (float*)alloc(ND * 4 * 4);
  float* x2     = (float*)alloc(ND * 4);
  float* x3     = (float*)alloc(ND * 4);
  float* ffnh   = (float*)alloc(ND * 4 * 4);
  unsigned short* pWq  = (unsigned short*)alloc(128 * 128 * 2);
  unsigned short* pWk  = (unsigned short*)alloc(128 * 128 * 2);
  unsigned short* pWv  = (unsigned short*)alloc(128 * 128 * 2);
  unsigned short* pWo  = (unsigned short*)alloc(128 * 128 * 2);
  unsigned short* pWe  = (unsigned short*)alloc(4 * 1 * 32 * 16 * 2);
  unsigned short* pWih = (unsigned short*)alloc(4 * 32 * 32 * 16 * 2);
  unsigned short* pWf1 = (unsigned short*)alloc(4 * 32 * 32 * 16 * 2);
  unsigned short* pWf2 = (unsigned short*)alloc(16 * 8 * 32 * 16 * 2);

  auto pack = [&](const float* W, unsigned short* P, int Kd, int Nn, int NT, int trans) {
    int total = (Kd >> 5) * NT * 32;
    pack_b_kernel<<<(total + 255) / 256, 256, 0, stream>>>(W, P, Kd, Nn, NT, trans);
  };
  pack(Wq, pWq, 128, 128, 8, 0);
  pack(Wk, pWk, 128, 128, 8, 0);
  pack(Wv, pWv, 128, 128, 8, 0);
  pack(Wo, pWo, 128, 128, 8, 0);
  pack(We, pWe, 128, 8, 1, 0);
  pack(Wih, pWih, 128, 512, 32, 1);   // x @ Wih^T : B[k,n] = Wih[n,k]
  pack(Wf1, pWf1, 128, 512, 32, 0);
  pack(Wf2, pWf2, 512, 128, 8, 0);

  auto gemm128 = [&](const float* A, const unsigned short* Bp, const float* bias,
                     const float* bias2, float* C, int M, int Nn, int NT, int act) {
    int tiles = (M >> 4) * NT;
    gemm_bf16_wmma<128><<<(tiles + 7) / 8, 256, 0, stream>>>(A, Bp, bias, bias2, C,
                                                             M, Nn, NT, act);
  };

  // QKV projections
  gemm128(h, pWq, bq, nullptr, qb,   NNODES, 128, 8, 0);
  gemm128(h, pWk, bk, nullptr, kbuf, NNODES, 128, 8, 0);
  gemm128(h, pWv, bv, nullptr, vb,   NNODES, 128, 8, 0);

  // Fused scores = e@We + be + q.k^T/4  (single 512 MB streaming pass)
  escore_kernel<<<((NNODES * NNODES) / 16 + 7) / 8, 256, 0, stream>>>(
      e, qb, kbuf, pWe, be, scores);

  // per-head top-k + softmax + attn.v
  topk_softmax_av_kernel<<<(NNODES * NHEAD) / 8, 256, 0, stream>>>(scores, vb, hattn);

  // output projection + residual + LN1
  gemm128(hattn, pWo, bo, nullptr, tmp, NNODES, 128, 8, 0);
  add_layernorm_kernel<<<NNODES, DDIM, 0, stream>>>(h, tmp, l1g, l1b, xb);

  // LSTM gates (zero hidden state: only bhh survives from the hh path)
  gemm128(xb, pWih, bih, bhh, gates, NNODES, 512, 32, 0);
  float* out0 = (float*)d_out;
  lstm_elem_kernel<<<(NNODES * DDIM) / 256, 256, 0, stream>>>(
      xb, gates, x2, out0 + ND, out0 + 2 * ND);

  // GraphNorm over nodes
  graphnorm_kernel<<<DDIM, 256, 0, stream>>>(x2, gng, gnb, x3);

  // FFN
  gemm128(x3, pWf1, bf1, nullptr, ffnh, NNODES, 512, 32, 1);   // ReLU fused
  {
    int tiles = (NNODES >> 4) * 8;
    gemm_bf16_wmma<512><<<(tiles + 7) / 8, 256, 0, stream>>>(
        ffnh, pWf2, bf2, nullptr, tmp, NNODES, 128, 8, 0);
  }

  // residual + LN2 -> out
  add_layernorm_kernel<<<NNODES, DDIM, 0, stream>>>(x3, tmp, l2g, l2b, out0);
}
